// DRAG_17892833755547
// MI455X (gfx1250) — matrix operand order repaired and checked by reading, not compiled, not run on hardware
//
#include <hip/hip_runtime.h>
#include <hip/hip_bf16.h>

// ---------------------------------------------------------------------------
// Types for CDNA5 WMMA
// ---------------------------------------------------------------------------
typedef unsigned short u16;
typedef __attribute__((ext_vector_type(16))) __bf16 v16bf;
typedef __attribute__((ext_vector_type(8)))  float  v8f;
typedef __attribute__((ext_vector_type(4)))  unsigned int u32x4;

union Frag16 { v16bf v; u32x4 u[2]; };

#if defined(__has_builtin)
#if __has_builtin(__builtin_amdgcn_global_load_async_to_lds_b128) && \
    __has_builtin(__builtin_amdgcn_s_wait_asynccnt)
#define HAVE_ASYNC_LDS 1
#endif
#endif

#if defined(HAVE_ASYNC_LDS)
// async-to-LDS builtin operand types: v4i in AS1 (global) / AS3 (shared)
typedef int v4i __attribute__((vector_size(16)));
typedef __attribute__((address_space(1))) v4i gas_v4i;
typedef __attribute__((address_space(3))) v4i las_v4i;
#endif

__device__ __forceinline__ u16 f2bf(float f) {
    unsigned u = __float_as_uint(f);
    unsigned r = (u + 0x7fffu + ((u >> 16) & 1u)) >> 16;   // round-to-nearest-even
    return (u16)r;
}
__device__ __forceinline__ float lrelu(float x, float s) {
    return x >= 0.f ? x : s * x;
}
// monotone float<->uint key for atomicMax-based segment max (memset 0 == -inf)
__device__ __forceinline__ unsigned fkey(float f) {
    unsigned i = __float_as_uint(f);
    return i ^ ((unsigned)((int)i >> 31) | 0x80000000u);
}
__device__ __forceinline__ float fdec(unsigned k) {
    unsigned i = (k & 0x80000000u) ? (k ^ 0x80000000u) : ~k;
    return __uint_as_float(i);
}

// ---------------------------------------------------------------------------
// Weight prep: fp32 row-major [K=128][N=128] -> bf16 transposed [N][K]
// ---------------------------------------------------------------------------
__global__ void prep_wt(const float* __restrict__ W, u16* __restrict__ Wt) {
    int n = blockIdx.x;      // 0..127
    int k = threadIdx.x;     // 0..127
    Wt[n * 128 + k] = f2bf(W[k * 128 + n]);
}

__global__ void to_bf16_k(const float* __restrict__ src, u16* __restrict__ dst, long n) {
    long i = (long)blockIdx.x * blockDim.x + threadIdx.x;
    if (i < n) dst[i] = f2bf(src[i]);
}

// ---------------------------------------------------------------------------
// GEMM: C[M x 128] = A[M x 128] (bf16) @ W[128 x 128] + bias, optional lrelu,
// optional f32 and/or bf16 outputs with independent leading dims.
// Wt is bf16 transposed (N-major). 256 threads = 8 waves; 128-row M tile.
// Weights staged in LDS via GLOBAL_LOAD_ASYNC_TO_LDS_B128 when available.
// Per K-step: preload 8 B fragments (16 ds_load_b128, pipelined), then issue
// 8 back-to-back v_wmma_f32_16x16x32_bf16.
// ---------------------------------------------------------------------------
__launch_bounds__(256)
__global__ void gemm128_bf16(const u16* __restrict__ A,
                             const u16* __restrict__ Wt,
                             const float* __restrict__ bias,
                             float* __restrict__ C,  int ldc,
                             u16*   __restrict__ Cb, int ldcb,
                             int M, int act, float slope) {
    __shared__ __align__(16) u16 sW[128 * 128];   // 32 KB

    int tid = threadIdx.x;
#if defined(HAVE_ASYNC_LDS)
    {   // async DMA of the 32 KB weight tile: 2048 x 16B, 8 chunks/thread
        #pragma unroll
        for (int t = 0; t < 8; ++t) {
            int idx = tid + t * 256;
            gas_v4i* g = (gas_v4i*)(uintptr_t)((const u32x4*)Wt + idx);
            las_v4i* l = (las_v4i*)(unsigned)(uintptr_t)((u32x4*)sW + idx);
            __builtin_amdgcn_global_load_async_to_lds_b128(g, l, 0, 0);
        }
        __builtin_amdgcn_s_wait_asynccnt(0);
    }
#else
    {   // fallback: stage via VGPRs
        const u32x4* s = (const u32x4*)Wt;
        u32x4* d = (u32x4*)sW;
        #pragma unroll
        for (int t = 0; t < 8; ++t) d[tid + t * 256] = s[tid + t * 256];
    }
#endif
    __syncthreads();

    int wave = tid >> 5, lane = tid & 31;
    int half = lane >> 4, l15 = lane & 15;
    long m0 = (long)blockIdx.x * 128;

    long arow = m0 + wave * 16 + l15;
    if (arow >= M) arow = M - 1;                  // clamp; stores are guarded
    const u16* Arow = A + arow * 128;

    v8f acc[8];
    #pragma unroll
    for (int nt = 0; nt < 8; ++nt)
        #pragma unroll
        for (int q = 0; q < 8; ++q) acc[nt][q] = 0.f;

    #pragma unroll
    for (int kk = 0; kk < 4; ++kk) {
        // A fragment: lane holds row (lane&15); two 8-K runs per ISA layout
        Frag16 a;
        a.u[0] = *(const u32x4*)(Arow + kk * 32 + half * 8);
        a.u[1] = *(const u32x4*)(Arow + kk * 32 + 16 + half * 8);
        // Preload all 8 B fragments for this K-step (pipelined ds_loads)
        Frag16 b[8];
        #pragma unroll
        for (int nt = 0; nt < 8; ++nt) {
            const u16* bp = sW + (nt * 16 + l15) * 128 + kk * 32 + half * 8;
            b[nt].u[0] = *(const u32x4*)(bp);
            b[nt].u[1] = *(const u32x4*)(bp + 16);
        }
        // Back-to-back WMMA chain
        #pragma unroll
        for (int nt = 0; nt < 8; ++nt)
            acc[nt] = __builtin_amdgcn_wmma_f32_16x16x32_bf16(
                false, a.v, false, b[nt].v, (short)0, acc[nt], false, false);
    }

    // Epilogue. C layout: VGPR r -> row r (+8 for lanes 16..31), col = lane&15.
    #pragma unroll
    for (int nt = 0; nt < 8; ++nt) {
        int col = nt * 16 + l15;
        float bv = bias ? bias[col] : 0.f;
        #pragma unroll
        for (int r = 0; r < 8; ++r) {
            long row = m0 + wave * 16 + r + half * 8;
            if (row < M) {
                float v = acc[nt][r] + bv;
                if (act) v = lrelu(v, slope);
                if (C)  C[row * (long)ldc + col] = v;
                if (Cb) Cb[row * (long)ldcb + col] = f2bf(v);
            }
        }
    }
}

// ---------------------------------------------------------------------------
// el/er per (node, head): dot over GD=32
// ---------------------------------------------------------------------------
__global__ void elr_k(const float* __restrict__ hW,
                      const float* __restrict__ al, const float* __restrict__ ar,
                      float* __restrict__ el, float* __restrict__ er, int n) {
    int t = blockIdx.x * blockDim.x + threadIdx.x;
    if (t >= n * 4) return;
    int node = t >> 2, g = t & 3;
    const float4* hp = (const float4*)(hW + (long)node * 128 + g * 32);
    const float4* ap = (const float4*)(al + g * 32);
    const float4* rp = (const float4*)(ar + g * 32);
    float sl = 0.f, sr = 0.f;
    #pragma unroll
    for (int q = 0; q < 8; ++q) {
        float4 h4 = hp[q], a4 = ap[q], r4 = rp[q];
        sl += h4.x * a4.x + h4.y * a4.y + h4.z * a4.z + h4.w * a4.w;
        sr += h4.x * r4.x + h4.y * r4.y + h4.z * r4.z + h4.w * r4.w;
    }
    el[t] = sl; er[t] = sr;
}

// ---------------------------------------------------------------------------
// Edge pass A: e = lrelu(el[src]+er[dst], 0.2); segment max via uint atomicMax
// ---------------------------------------------------------------------------
__global__ void edgeA_k(const int* __restrict__ src, const int* __restrict__ dst,
                        const float* __restrict__ el, const float* __restrict__ er,
                        float* __restrict__ eBuf, unsigned* __restrict__ emax, int nE) {
    int e = blockIdx.x * blockDim.x + threadIdx.x;
    if (e >= nE) return;
    int s = src[e], d = dst[e];
    #pragma unroll
    for (int g = 0; g < 4; ++g) {
        float v = lrelu(el[s * 4 + g] + er[d * 4 + g], 0.2f);
        eBuf[e * 4 + g] = v;
        atomicMax(&emax[d * 4 + g], fkey(v));
    }
}

// Edge pass B: ee = exp(e - emax[dst]); den[dst] += ee
__global__ void edgeB_k(const int* __restrict__ dst,
                        float* __restrict__ eBuf, const unsigned* __restrict__ emax,
                        float* __restrict__ den, int nE) {
    int e = blockIdx.x * blockDim.x + threadIdx.x;
    if (e >= nE) return;
    int d = dst[e];
    #pragma unroll
    for (int g = 0; g < 4; ++g) {
        float m = fdec(emax[d * 4 + g]);
        float ee = __expf(eBuf[e * 4 + g] - m);
        eBuf[e * 4 + g] = ee;
        atomicAdd(&den[d * 4 + g], ee);
    }
}

// Edge pass C: one wave per edge; feats[dst, slot, :] += a * hW[src, :]
__global__ void edgeC_k(const int* __restrict__ src, const int* __restrict__ dst,
                        const float* __restrict__ eBuf, const float* __restrict__ den,
                        const float* __restrict__ hW, float* __restrict__ feats,
                        int nE, int slot) {
    int gt = blockIdx.x * blockDim.x + threadIdx.x;
    int e = gt >> 5, lane = gt & 31;
    if (e >= nE) return;
    int s = src[e], d = dst[e];
    int c0 = lane * 4, g = lane >> 3;           // head = col/32
    float a = eBuf[e * 4 + g] / den[d * 4 + g];
    const float* hp = hW + (long)s * 128 + c0;
    __builtin_prefetch(hp, 0, 0);
    float4 h4 = *(const float4*)hp;
    float* out = feats + (long)d * 512 + slot * 128 + c0;
    atomicAdd(out + 0, a * h4.x);
    atomicAdd(out + 1, a * h4.y);
    atomicAdd(out + 2, a * h4.z);
    atomicAdd(out + 3, a * h4.w);
}

// feats: apply lrelu(0.01) and emit bf16 copy for the attention GEMMs
__global__ void featsFin_k(const float* __restrict__ feats, u16* __restrict__ featsB, long n) {
    long i = (long)blockIdx.x * blockDim.x + threadIdx.x;
    if (i < n) featsB[i] = f2bf(lrelu(feats[i], 0.01f));
}

// ---------------------------------------------------------------------------
// Gated multi-head attention: thread per (node, head a of 8), HID=16, k<=4
// ---------------------------------------------------------------------------
__global__ void attn_k(const float* __restrict__ gl, const float* __restrict__ gr,
                       const float* __restrict__ vv,
                       const float* __restrict__ aW, const float* __restrict__ ab,
                       int n, int k,
                       float* __restrict__ outF, u16* __restrict__ outB,
                       u16* __restrict__ lfB, int lfStride, int lfSlot, int nLf) {
    int t = blockIdx.x * blockDim.x + threadIdx.x;
    if (t >= n * 8) return;
    int node = t >> 3, a = t & 7;

    float g16[16], aw[16];
    const float* glp = gl + (long)node * 128 + a * 16;
    #pragma unroll
    for (int h = 0; h < 16; ++h) { g16[h] = glp[h]; aw[h] = aW[h]; }
    float abv = ab[0];

    float ek[4];
    for (int kk = 0; kk < k; ++kk) {
        const float* grp = gr + ((long)node * k + kk) * 128 + a * 16;
        float s = 0.f;
        #pragma unroll
        for (int h = 0; h < 16; ++h) s += lrelu(g16[h] + grp[h], 0.01f) * aw[h];
        ek[kk] = s + abv;
    }
    float m = ek[0];
    for (int kk = 1; kk < k; ++kk) m = fmaxf(m, ek[kk]);
    float ssum = 0.f;
    for (int kk = 0; kk < k; ++kk) { ek[kk] = __expf(ek[kk] - m); ssum += ek[kk]; }
    float inv = 1.f / ssum;

    float o[16];
    #pragma unroll
    for (int h = 0; h < 16; ++h) o[h] = 0.f;
    for (int kk = 0; kk < k; ++kk) {
        float p = ek[kk] * inv;
        const float* vp = vv + ((long)node * k + kk) * 128 + a * 16;
        #pragma unroll
        for (int h = 0; h < 16; ++h) o[h] += p * vp[h];
    }

    long base = (long)node * 128 + a * 16;
    #pragma unroll
    for (int h = 0; h < 16; ++h) {
        float x = lrelu(o[h], 0.01f);
        if (outF) outF[base + h] = x;
        if (outB) outB[base + h] = f2bf(x);
        if (lfB && node < nLf)
            lfB[(long)node * lfStride + lfSlot * 128 + a * 16 + h] = f2bf(x);
    }
}

// Classifier: thread per (node, class); out = h @ lin_W + lin_b (NC = 2)
__global__ void class_k(const float* __restrict__ h, const float* __restrict__ linW,
                        const float* __restrict__ linb, float* __restrict__ out, int n) {
    int t = blockIdx.x * blockDim.x + threadIdx.x;
    if (t >= n * 2) return;
    int node = t >> 1, c = t & 1;
    const float* hp = h + (long)node * 128;
    float s = linb[c];
    #pragma unroll 4
    for (int k = 0; k < 128; ++k) s += hp[k] * linW[k * 2 + c];
    out[t] = s;
}

// ---------------------------------------------------------------------------
// Host orchestration
// ---------------------------------------------------------------------------
static inline int cdiv(long a, long b) { return (int)((a + b - 1) / b); }

extern "C" void kernel_launch(void* const* d_in, const int* in_sizes, int n_in,
                              void* d_out, int out_size, void* d_ws, size_t ws_size,
                              hipStream_t stream) {
    (void)in_sizes; (void)n_in; (void)out_size; (void)ws_size;
    const int NN[3] = {100000, 50000, 25000};
    const int EE[2] = {500000, 250000};
    const int NOUT = NN[2];

    const float* x       = (const float*)d_in[0];
    const int*   srcA[2] = {(const int*)d_in[1], (const int*)d_in[3]};
    const int*   dstA[2] = {(const int*)d_in[2], (const int*)d_in[4]};
    const float* conv_al = (const float*)d_in[6];
    const float* conv_ar = (const float*)d_in[7];
    const float* feat_b  = (const float*)d_in[9];
    const float* relWl_b = (const float*)d_in[11];
    const float* relWr_b = (const float*)d_in[13];
    const float* relP_b  = (const float*)d_in[15];
    const float* rela_W  = (const float*)d_in[16];
    const float* rela_b  = (const float*)d_in[17];
    const float* proj_b  = (const float*)d_in[19];
    const float* hopWl_b = (const float*)d_in[21];
    const float* hopWr_b = (const float*)d_in[23];
    const float* hopP_b  = (const float*)d_in[25];
    const float* hopa_W  = (const float*)d_in[26];
    const float* hopa_b  = (const float*)d_in[27];
    const float* lin_W   = (const float*)d_in[28];
    const float* lin_b   = (const float*)d_in[29];

    // ---- workspace carve-out -------------------------------------------
    char* p = (char*)d_ws;
    auto carve = [&](size_t bytes) -> void* {
        void* r = (void*)p;
        p += (bytes + 255) & ~(size_t)255;
        return r;
    };
    const size_t MAT = 128 * 128;
    u16*      WT     = (u16*)     carve(18 * MAT * sizeof(u16));
    u16*      xB     = (u16*)     carve((size_t)NN[0] * 128 * sizeof(u16));
    u16*      hB     = (u16*)     carve((size_t)NN[1] * 128 * sizeof(u16));
    u16*      h2B    = (u16*)     carve((size_t)NN[2] * 128 * sizeof(u16));
    float*    hW     = (float*)   carve((size_t)NN[0] * 128 * sizeof(float));
    float*    el     = (float*)   carve((size_t)NN[0] * 4 * sizeof(float));
    float*    er     = (float*)   carve((size_t)NN[0] * 4 * sizeof(float));
    float*    eBuf   = (float*)   carve((size_t)EE[0] * 4 * sizeof(float));
    unsigned* emaxU  = (unsigned*)carve((size_t)NN[1] * 4 * sizeof(unsigned));
    float*    den    = (float*)   carve((size_t)NN[1] * 4 * sizeof(float));
    float*    feats  = (float*)   carve((size_t)NN[1] * 512 * sizeof(float));
    u16*      featsB = (u16*)     carve((size_t)NN[1] * 512 * sizeof(u16));
    float*    gl     = (float*)   carve((size_t)NN[1] * 128 * sizeof(float));
    float*    gr     = (float*)   carve((size_t)NN[1] * 4 * 128 * sizeof(float));
    float*    vv     = (float*)   carve((size_t)NN[1] * 4 * 128 * sizeof(float));
    u16*      lfB    = (u16*)     carve((size_t)NOUT * 384 * sizeof(u16));
    float*    outF   = (float*)   carve((size_t)NOUT * 128 * sizeof(float));

    // ---- weight prep: 18 matrices -> bf16 transposed -------------------
    struct { const float* src; int cnt; } wl[9] = {
        {(const float*)d_in[5],  6},   // conv_W   -> WT[0..5]
        {(const float*)d_in[8],  2},   // feat_W   -> WT[6..7]
        {(const float*)d_in[10], 2},   // relWl_W  -> WT[8..9]
        {(const float*)d_in[12], 2},   // relWr_W  -> WT[10..11]
        {(const float*)d_in[14], 2},   // relP_W   -> WT[12..13]
        {(const float*)d_in[18], 1},   // proj_W   -> WT[14]
        {(const float*)d_in[20], 1},   // hopWl_W  -> WT[15]
        {(const float*)d_in[22], 1},   // hopWr_W  -> WT[16]
        {(const float*)d_in[24], 1},   // hopP_W   -> WT[17]
    };
    int wi = 0;
    for (int e = 0; e < 9; ++e)
        for (int m = 0; m < wl[e].cnt; ++m, ++wi)
            prep_wt<<<128, 128, 0, stream>>>(wl[e].src + (size_t)m * MAT, WT + (size_t)wi * MAT);

    to_bf16_k<<<cdiv((long)NN[0] * 128, 256), 256, 0, stream>>>(x, xB, (long)NN[0] * 128);

    auto gemm = [&](const u16* A, int wtIdx, const float* bias,
                    float* C, int ldc, u16* Cb, int ldcb, int M, int act) {
        gemm128_bf16<<<cdiv(M, 128), 256, 0, stream>>>(
            A, WT + (size_t)wtIdx * MAT, bias, C, ldc, Cb, ldcb, M, act, 0.01f);
    };

    // ---- layers ---------------------------------------------------------
    const u16* hIn = xB;
    for (int i = 0; i < 2; ++i) {
        int nS = NN[i], nD = NN[i + 1], nE = EE[i];
        (void)hipMemsetAsync(feats, 0, (size_t)nD * 512 * sizeof(float), stream);
        for (int j = 0; j < 3; ++j) {
            gemm(hIn, i * 3 + j, nullptr, hW, 128, nullptr, 0, nS, 0);
            elr_k<<<cdiv((long)nS * 4, 256), 256, 0, stream>>>(
                hW, conv_al + (size_t)(i * 3 + j) * 128, conv_ar + (size_t)(i * 3 + j) * 128,
                el, er, nS);
            (void)hipMemsetAsync(emaxU, 0, (size_t)nD * 4 * sizeof(unsigned), stream);
            (void)hipMemsetAsync(den,   0, (size_t)nD * 4 * sizeof(float),    stream);
            const int* sj = srcA[i] + (size_t)j * nE;
            const int* dj = dstA[i] + (size_t)j * nE;
            edgeA_k<<<cdiv(nE, 256), 256, 0, stream>>>(sj, dj, el, er, eBuf, emaxU, nE);
            edgeB_k<<<cdiv(nE, 256), 256, 0, stream>>>(dj, eBuf, emaxU, den, nE);
            edgeC_k<<<cdiv((long)nE * 32, 256), 256, 0, stream>>>(
                sj, dj, eBuf, den, hW, feats, nE, j);
        }
        // feats slot 3 = x_dst @ feat_W + b (lrelu applied in featsFin)
        gemm(hIn, 6 + i, feat_b + (size_t)i * 128, feats + 384, 512, nullptr, 0, nD, 0);
        featsFin_k<<<cdiv((long)nD * 512, 256), 256, 0, stream>>>(feats, featsB, (long)nD * 512);
        // gated attention GEMMs
        gemm(hIn,    8 + i,  relWl_b + (size_t)i * 128, gl, 128, nullptr, 0, nD, 0);
        gemm(featsB, 10 + i, relWr_b + (size_t)i * 128, gr, 128, nullptr, 0, nD * 4, 0);
        gemm(featsB, 12 + i, relP_b  + (size_t)i * 128, vv, 128, nullptr, 0, nD * 4, 0);
        u16* outB = (i == 0) ? hB : h2B;
        attn_k<<<cdiv((long)nD * 8, 256), 256, 0, stream>>>(
            gl, gr, vv, rela_W + (size_t)i * 16, rela_b + i, nD, 4,
            nullptr, outB, lfB, 384, i + 1, NOUT);
        hIn = outB;
    }

    // ---- hop attention + classifier ------------------------------------
    gemm(xB, 14, proj_b, nullptr, 0, lfB, 384, NOUT, 0);        // layer_features[0]
    gemm(xB,  15, hopWl_b, gl, 128, nullptr, 0, NOUT, 0);
    gemm(lfB, 16, hopWr_b, gr, 128, nullptr, 0, NOUT * 3, 0);
    gemm(lfB, 17, hopP_b,  vv, 128, nullptr, 0, NOUT * 3, 0);
    attn_k<<<cdiv((long)NOUT * 8, 256), 256, 0, stream>>>(
        gl, gr, vv, hopa_W, hopa_b, NOUT, 3,
        outF, nullptr, nullptr, 0, 0, 0);
    class_k<<<cdiv((long)NOUT * 2, 256), 256, 0, stream>>>(
        outF, lin_W, lin_b, (float*)d_out, NOUT);
}
